// ProbAttention_309237645724
// MI455X (gfx1250) — compile-verified, matrix operand order checked by reading, not run on hardware
//
#include <hip/hip_runtime.h>
#include <hip/hip_bf16.h>
#include <cstdint>

typedef __attribute__((ext_vector_type(16))) _Float16 v16h;
typedef __attribute__((ext_vector_type(8)))  float    v8f;
typedef __attribute__((__vector_size__(16))) int      v4i;

constexpr int Ln   = 2048;   // sequence length L == S
constexpr int Dn   = 64;     // head dim
constexpr int BH   = 16;     // B*H
constexpr int NTOP = 40;     // 5 * ceil(ln(2048))
constexpr int MT   = 3;      // ceil(40/16) row tiles
constexpr int MP   = 48;     // padded row count (MT*16)
constexpr int NT   = Ln / 16;
constexpr int KT   = Ln / 32;   // k-steps for attn*V
#define NEG_INF (-__builtin_huge_valf())

// ---- gfx1250 async global->LDS path (guarded; falls back to sync staging) ----
#if defined(__HIP_DEVICE_COMPILE__) && __has_builtin(__builtin_amdgcn_global_load_async_to_lds_b128)
#define ASYNC_LDS 1
#endif

#if defined(ASYNC_LDS)
#if __has_builtin(__builtin_amdgcn_s_wait_asynccnt)
#define WAIT_ASYNC(n) __builtin_amdgcn_s_wait_asynccnt(n)
#else
#define WAIT_ASYNC(n) asm volatile("s_wait_asynccnt %0" ::"i"(n) : "memory")
#endif
#else
#define WAIT_ASYNC(n) ((void)0)
#endif

#if defined(ASYNC_LDS)
typedef __attribute__((address_space(1))) v4i gv4i;   // global int4
typedef __attribute__((address_space(3))) v4i lv4i;   // LDS int4
#endif

__device__ __forceinline__ void stage16(void* lds, const void* g) {
#if defined(ASYNC_LDS)
    // per-lane 16B DMA: global (as1) -> LDS (as3); flat LDS ptr low 32b == LDS offset
    __builtin_amdgcn_global_load_async_to_lds_b128(
        (gv4i*)(uintptr_t)g,
        (lv4i*)(unsigned)(uintptr_t)lds,
        0, 0);
#else
    *(float4*)lds = *(const float4*)g;
#endif
}

// K-index of element e (VGPR e/2, half e%2) in the 16x16x32 f16 A/B layout,
// per ISA 7.12.2: lanes 0-15 -> K {0..7,16..23}, lanes 16-31 -> +8.
__device__ __forceinline__ int wmma_k_of(int e, int group) {
    int v = e >> 1, p = e & 1;
    int k = (v < 4) ? (v * 2 + p) : (16 + (v - 4) * 2 + p);
    return k + group * 8;
}

// ---------------- 1) sparsity measure M = max_u - sum_u / S ----------------
__global__ __launch_bounds__(256) void k_measure(const float* __restrict__ Q,
                                                 const float* __restrict__ K,
                                                 const int* __restrict__ samp,
                                                 float* __restrict__ Mv) {
    int gid  = blockIdx.x * blockDim.x + threadIdx.x;
    int wid  = gid >> 5, lane = gid & 31;
    int bh   = wid / Ln, l = wid % Ln;
    const float2 q = ((const float2*)(Q + (size_t)(bh * Ln + l) * Dn))[lane];
    float mx = NEG_INF, sm = 0.0f;
    for (int u = 0; u < NTOP; ++u) {
        int ki = samp[l * NTOP + u];
        const float2 kv = ((const float2*)(K + (size_t)(bh * Ln + ki) * Dn))[lane];
        float p = q.x * kv.x + q.y * kv.y;
        #pragma unroll
        for (int off = 16; off; off >>= 1) p += __shfl_xor(p, off);
        mx = fmaxf(mx, p);
        sm += p;
    }
    if (lane == 0) Mv[(size_t)bh * Ln + l] = mx - sm * (1.0f / (float)Ln);
}

// ---------------- 2) top-40 per (b,h): iterative argmax ----------------
__global__ __launch_bounds__(256) void k_topk(const float* __restrict__ Mv,
                                              int* __restrict__ Mtop) {
    __shared__ float sv[Ln];
    __shared__ float rv[256];
    __shared__ int   ri[256];
    int bh = blockIdx.x, tid = threadIdx.x;
    for (int i = tid; i < Ln; i += 256) sv[i] = Mv[(size_t)bh * Ln + i];
    __syncthreads();
    for (int j = 0; j < NTOP; ++j) {
        float best = NEG_INF; int bi = 0x7fffffff;
        for (int i = tid; i < Ln; i += 256) {
            float v = sv[i];
            if (v > best || (v == best && i < bi)) { best = v; bi = i; }
        }
        rv[tid] = best; ri[tid] = bi;
        __syncthreads();
        for (int s = 128; s > 0; s >>= 1) {
            if (tid < s) {
                if (rv[tid + s] > rv[tid] ||
                    (rv[tid + s] == rv[tid] && ri[tid + s] < ri[tid])) {
                    rv[tid] = rv[tid + s]; ri[tid] = ri[tid + s];
                }
            }
            __syncthreads();
        }
        if (tid == 0) { Mtop[bh * NTOP + j] = ri[0]; sv[ri[0]] = NEG_INF; }
        __syncthreads();
    }
}

// ------- 3) scores = Q_top * K^T * scale, causal-masked (padded 48 rows) -------
__global__ __launch_bounds__(32) void k_scores(const float* __restrict__ Q,
                                               const float* __restrict__ K,
                                               const int* __restrict__ Mtop,
                                               float* __restrict__ scores) {
    int tile  = blockIdx.x;            // BH * MT * NT tiles
    int ntile = tile % NT; tile /= NT;
    int mtile = tile % MT;
    int bh    = tile / MT;
    int lane  = threadIdx.x, lr = lane & 15, group = lane >> 4;

    int r  = mtile * 16 + lr;                       // query-tile row (0..47)
    int qi = (r < NTOP) ? Mtop[bh * NTOP + r] : 0;  // pad rows read row 0
    const float* qrow = Q + (size_t)(bh * Ln + qi) * Dn;
    int s = ntile * 16 + lr;                        // key column
    const float* krow = K + (size_t)(bh * Ln + s) * Dn;

    v8f acc = {};
    #pragma unroll
    for (int kk = 0; kk < 2; ++kk) {                // K = 64 = 2 x 32
        v16h a, b;
        #pragma unroll
        for (int e = 0; e < 16; ++e) {
            int k = wmma_k_of(e, group) + kk * 32;
            a[e] = (_Float16)qrow[k];
            b[e] = (_Float16)krow[k];
        }
        acc = __builtin_amdgcn_wmma_f32_16x16x32_f16(
            false, a, false, b, (short)0, acc, false, false);
    }
    #pragma unroll
    for (int rr = 0; rr < 8; ++rr) {
        int m = mtile * 16 + rr + 8 * group;
        // causal threshold for row m is qi held by lane (rr + 8*group)
        int thr = __shfl(qi, rr + 8 * group, 32);
        float val = acc[rr] * 0.125f;               // 1/sqrt(64)
        if (s > thr) val = NEG_INF;
        scores[((size_t)bh * MP + m) * Ln + s] = val;   // unconditional store
    }
}

// ---------------- 4) row softmax over S=2048 (real rows only) ----------------
__global__ __launch_bounds__(256) void k_softmax(float* __restrict__ scores) {
    __shared__ float red[256];
    int row = blockIdx.x, tid = threadIdx.x;        // BH*NTOP rows
    int bh = row / NTOP, u = row % NTOP;
    float* p = scores + ((size_t)bh * MP + u) * Ln;
    float mx = NEG_INF;
    for (int i = tid; i < Ln; i += 256) mx = fmaxf(mx, p[i]);
    red[tid] = mx; __syncthreads();
    for (int s = 128; s > 0; s >>= 1) {
        if (tid < s) red[tid] = fmaxf(red[tid], red[tid + s]);
        __syncthreads();
    }
    mx = red[0]; __syncthreads();
    float sm = 0.0f;
    for (int i = tid; i < Ln; i += 256) {
        float e = __expf(p[i] - mx);   // exp(-inf - mx) == 0 handles the mask
        p[i] = e; sm += e;
    }
    red[tid] = sm; __syncthreads();
    for (int s = 128; s > 0; s >>= 1) {
        if (tid < s) red[tid] += red[tid + s];
        __syncthreads();
    }
    float inv = 1.0f / red[0];
    for (int i = tid; i < Ln; i += 256) p[i] *= inv;
}

// ------- 5) upd = attn * V: LDS-staged, async double-buffered, 4 waves -------
__global__ __launch_bounds__(128) void k_av(const float* __restrict__ scores,
                                            const float* __restrict__ V,
                                            float* __restrict__ upd) {
    __shared__ float Vt[2][32 * Dn];    // 32 s-rows x 64 d-cols (8 KB each)
    __shared__ float At[2][16 * 32];    // 16 m-rows x 32 s-cols (2 KB each)

    int blk   = blockIdx.x;             // BH * MT blocks
    int mtile = blk % MT;
    int bh    = blk / MT;
    int tid   = threadIdx.x;
    int lane  = tid & 31, wave = tid >> 5;     // wave == d-tile (0..3)
    int lr    = lane & 15, group = lane >> 4;

    const float* Vbase = V + (size_t)bh * Ln * Dn;
    const float* Abase = scores + ((size_t)bh * MP + mtile * 16) * Ln;

    int arr = tid >> 3, akc = (tid & 7) * 4;   // A staging: row, col4

    auto stage = [&](int kt, int buf) {
        const float* vsrc = Vbase + (size_t)kt * 32 * Dn;
        #pragma unroll
        for (int i = 0; i < 4; ++i) {
            int c4 = (tid + 128 * i) * 4;      // float index within 32x64 tile
            stage16(&Vt[buf][c4], vsrc + c4);
        }
        stage16(&At[buf][arr * 32 + akc],
                Abase + (size_t)arr * Ln + kt * 32 + akc);
    };

    int dcol = wave * 16 + lr;
    v8f acc = {};
    stage(0, 0);
    for (int kt = 0; kt < KT; ++kt) {
        int buf = kt & 1;
        if (kt + 1 < KT) { stage(kt + 1, buf ^ 1); WAIT_ASYNC(5); }
        else             { WAIT_ASYNC(0); }
        __syncthreads();                        // stage kt visible to all waves
        v16h a, b;
        #pragma unroll
        for (int e = 0; e < 16; ++e) {
            int k = wmma_k_of(e, group);        // s within 32-chunk
            a[e] = (_Float16)At[buf][lr * 32 + k];
            b[e] = (_Float16)Vt[buf][k * Dn + dcol];
        }
        acc = __builtin_amdgcn_wmma_f32_16x16x32_f16(
            false, a, false, b, (short)0, acc, false, false);
        __syncthreads();                        // done reading buf before reuse
    }
    #pragma unroll
    for (int rr = 0; rr < 8; ++rr) {
        int mm = mtile * 16 + rr + 8 * group;
        upd[((size_t)bh * MP + mm) * Dn + dcol] = acc[rr];
    }
}

// ---------------- 6) base output: cumsum(V) over the D axis ----------------
__global__ __launch_bounds__(256) void k_cumsum(const float* __restrict__ V,
                                                float* __restrict__ out) {
    int row = blockIdx.x * blockDim.x + threadIdx.x;   // 0..BH*Ln-1
    if (row >= BH * Ln) return;
    const float4* vr = (const float4*)(V + (size_t)row * Dn);
    float4* orow = (float4*)(out + (size_t)row * Dn);
    float run = 0.0f;
    #pragma unroll
    for (int i = 0; i < Dn / 4; ++i) {
        float4 x = vr[i];
        x.x += run; x.y += x.x; x.z += x.y; x.w += x.z; run = x.w;
        orow[i] = x;
    }
}

// ---------------- 7) scatter upd rows into output ----------------
__global__ __launch_bounds__(256) void k_scatter(const int* __restrict__ Mtop,
                                                 const float* __restrict__ upd,
                                                 float* __restrict__ out) {
    int gid = blockIdx.x * blockDim.x + threadIdx.x;   // BH*NTOP*Dn
    if (gid >= BH * NTOP * Dn) return;
    int d = gid % Dn, row = gid / Dn;                  // row = bh*NTOP + u
    int bh = row / NTOP, u = row % NTOP;
    int s = Mtop[row];
    out[(size_t)(bh * Ln + s) * Dn + d] = upd[((size_t)bh * MP + u) * Dn + d];
}

extern "C" void kernel_launch(void* const* d_in, const int* in_sizes, int n_in,
                              void* d_out, int out_size, void* d_ws, size_t ws_size,
                              hipStream_t stream) {
    (void)in_sizes; (void)n_in; (void)out_size; (void)ws_size;
    const float* Q   = (const float*)d_in[0];
    const float* K   = (const float*)d_in[1];
    const float* V   = (const float*)d_in[2];
    const int* samp  = (const int*)d_in[3];
    float* out       = (float*)d_out;

    char* ws      = (char*)d_ws;
    float* Mv     = (float*)ws;                          // BH*Ln floats (128 KB)
    int*   Mtop   = (int*)(ws + 131072);                 // BH*NTOP ints
    float* scores = (float*)(ws + 135168);               // BH*48*Ln floats (6 MB)
    float* upd    = (float*)(ws + 135168 + (size_t)BH * MP * Ln * 4);  // BH*48*Dn

    k_measure<<<(BH * Ln * 32) / 256, 256, 0, stream>>>(Q, K, samp, Mv);
    k_topk<<<BH, 256, 0, stream>>>(Mv, Mtop);
    k_scores<<<BH * MT * NT, 32, 0, stream>>>(Q, K, Mtop, scores);
    k_softmax<<<BH * NTOP, 256, 0, stream>>>(scores);
    k_av<<<BH * MT, 128, 0, stream>>>(scores, V, upd);
    k_cumsum<<<(BH * Ln + 255) / 256, 256, 0, stream>>>(V, out);
    k_scatter<<<(BH * NTOP * Dn + 255) / 256, 256, 0, stream>>>(Mtop, upd, out);
}
